// MoEAttention_15204184227983
// MI455X (gfx1250) — compile-verified, matrix operand order checked by reading, not run on hardware
//
#include <hip/hip_runtime.h>

// ---------------------------------------------------------------------------
// MoE span attention for MI455X (gfx1250, wave32, WMMA 16x16x32 f16->f32).
//
// Sizes (fixed by the reference): B=4, T=2048, C=1024, E=4 spans==experts,
// H=16 heads, Dh=64, S=512, M = B*T = 8192 rows, K=N=1024.
//
// Pipeline (all on `stream`):
//   1) convert hidden f32 -> f16 (ws)
//   2) gate: mean-pool span -> MLP -> softmax -> gw[16][4]; eff biases
//   3) q/k/v GEMMs: Y = X @ (sum_e gw_e W_e) + b_eff   (gating folded into
//      the weight-tile load; expert weights are L2-resident so the 4x
//      re-read is cheap). A-tile staged via GLOBAL_LOAD_ASYNC_TO_LDS,
//      overlapping with the VALU gating combine; waited with ASYNCcnt.
//   4) per-(inst,head) attention with full-row softmax in LDS
//   5) output GEMM (same kernel, E=1) -> d_out f32
// ---------------------------------------------------------------------------

typedef __attribute__((ext_vector_type(16))) _Float16 v16h;
typedef __attribute__((ext_vector_type(8)))  _Float16 v8h;
typedef __attribute__((ext_vector_type(8)))  float    v8f;
typedef __attribute__((ext_vector_type(4)))  float    v4f;
typedef __attribute__((ext_vector_type(4)))  int      v4i;

#if __has_builtin(__builtin_amdgcn_global_load_async_to_lds_b128)
#define ASYNC_LDS 1
typedef __attribute__((address_space(1))) v4i* gas_v4i_ptr;
typedef __attribute__((address_space(3))) v4i* las_v4i_ptr;
#endif

__device__ __forceinline__ void wait_asynccnt0() {
#if defined(ASYNC_LDS)
#if __has_builtin(__builtin_amdgcn_s_wait_asynccnt)
  __builtin_amdgcn_s_wait_asynccnt(0);
#else
  asm volatile("s_wait_asynccnt 0" ::: "memory");
#endif
#endif
}

__device__ __forceinline__ v8f wmma32(v16h a, v16h b, v8f c) {
  // (neg_a, A, neg_b, B, c_mod, C, reuse_a, reuse_b)
  return __builtin_amdgcn_wmma_f32_16x16x32_f16(false, a, false, b, (short)0, c,
                                                false, false);
}

// Load one 16x32 (A-style) / 32x16 (B-style) f16 WMMA fragment.
// Caller passes p = &buf[(lane%16)*stride + (lane/16)*8 (+k_base)], 16B aligned.
// Per ISA 7.12.2: halves 0..7 = K at p[0..7], halves 8..15 = K at p[16..23].
__device__ __forceinline__ v16h frag_ld(const _Float16* p) {
  v8h lo = *(const v8h*)p;
  v8h hi = *(const v8h*)(p + 16);
  v16h r;
#pragma unroll
  for (int j = 0; j < 8; ++j) { r[j] = lo[j]; r[j + 8] = hi[j]; }
  return r;
}

// ------------------------------- constants ---------------------------------
#define MDIM 8192
#define KDIM 1024
#define NDIM 1024
#define NINST 16
#define SPAN 512

// workspace layout (bytes)
#define OFF_XH  ((size_t)0)
#define OFF_Q   ((size_t)16 << 20)
#define OFF_K   ((size_t)32 << 20)
#define OFF_V   ((size_t)48 << 20)
#define OFF_AO  ((size_t)64 << 20)
#define OFF_GW  ((size_t)80 << 20)
#define OFF_BE  (OFF_GW + 4096)

// --------------------------- 1) f32 -> f16 ---------------------------------
__global__ void cvt_f32_to_f16(const float* __restrict__ x,
                               _Float16* __restrict__ y, int n) {
  int n4 = n >> 2;
  for (int i = blockIdx.x * blockDim.x + threadIdx.x; i < n4;
       i += gridDim.x * blockDim.x) {
    v4f v = *(const v4f*)(x + 4 * (size_t)i);
    _Float16* d = y + 4 * (size_t)i;
    d[0] = (_Float16)v[0]; d[1] = (_Float16)v[1];
    d[2] = (_Float16)v[2]; d[3] = (_Float16)v[3];
  }
}

// ------------------------------- 2) gate -----------------------------------
__global__ void gate_kernel(const float* __restrict__ X,
                            const float* __restrict__ gW1,
                            const float* __restrict__ gb1,
                            const float* __restrict__ gW2,
                            const float* __restrict__ gb2,
                            const float* __restrict__ bq,
                            const float* __restrict__ bk,
                            const float* __restrict__ bv,
                            float* __restrict__ gw_out,
                            float* __restrict__ beff) {
  __shared__ float pooled[1024];
  __shared__ float gact[128];
  __shared__ float lg[4];
  __shared__ float gws[4];
  const int inst = blockIdx.x;
  const int tid = threadIdx.x;

  for (int c = tid; c < 1024; c += 256) {
    float s = 0.f;
    for (int r = 0; r < SPAN; ++r)
      s += X[((size_t)inst * SPAN + r) * 1024 + c];
    pooled[c] = s * (1.0f / (float)SPAN);
  }
  __syncthreads();

  if (tid < 128) {
    float s = gb1[tid];
    for (int c = 0; c < 1024; ++c) s += pooled[c] * gW1[c * 128 + tid];
    gact[tid] = fmaxf(s, 0.f);
  }
  __syncthreads();

  if (tid < 4) {
    float s = gb2[tid];
    for (int hh = 0; hh < 128; ++hh) s += gact[hh] * gW2[hh * 4 + tid];
    lg[tid] = s;
  }
  __syncthreads();

  if (tid == 0) {
    float mx = fmaxf(fmaxf(lg[0], lg[1]), fmaxf(lg[2], lg[3]));
    float e0 = __expf(lg[0] - mx), e1 = __expf(lg[1] - mx);
    float e2 = __expf(lg[2] - mx), e3 = __expf(lg[3] - mx);
    float inv = 1.0f / (e0 + e1 + e2 + e3);
    gws[0] = e0 * inv; gws[1] = e1 * inv; gws[2] = e2 * inv; gws[3] = e3 * inv;
    gw_out[inst * 4 + 0] = gws[0]; gw_out[inst * 4 + 1] = gws[1];
    gw_out[inst * 4 + 2] = gws[2]; gw_out[inst * 4 + 3] = gws[3];
  }
  __syncthreads();

  for (int d = tid; d < 1024; d += 256) {
    float a0 = 0.f, a1 = 0.f, a2 = 0.f;
#pragma unroll
    for (int e = 0; e < 4; ++e) {
      float g = gws[e];
      a0 += g * bq[e * 1024 + d];
      a1 += g * bk[e * 1024 + d];
      a2 += g * bv[e * 1024 + d];
    }
    beff[(0 * NINST + inst) * 1024 + d] = a0;
    beff[(1 * NINST + inst) * 1024 + d] = a1;
    beff[(2 * NINST + inst) * 1024 + d] = a2;
  }
}

// ---------------------- 3/5) WMMA GEMM w/ gating combine -------------------
// grid = (N/128, M/128), block = 256 (8 waves, 2x4 wave grid, 64x32/wave).
template <typename OutT, bool MOE>
__global__ void moe_gemm(const _Float16* __restrict__ A,  // [M][K] f16
                         const float* __restrict__ W,     // [E][K][N] f32
                         const float* __restrict__ gw,    // [NINST][E]
                         const float* __restrict__ bias,  // MOE:[NINST][N] else [N]
                         OutT* __restrict__ Y, int E) {
  __shared__ _Float16 As[128 * 40] __attribute__((aligned(16)));  // pad 40
  __shared__ _Float16 Ws[128 * 40] __attribute__((aligned(16)));  // [n][k] pad

  const int tid = threadIdx.x;
  const int lane = tid & 31, wave = tid >> 5;
  const int rl = lane & 15;
  const int sel8 = (lane >> 4) * 8;
  const int row0 = blockIdx.y * 128;
  const int n0 = blockIdx.x * 128;
  const int inst = row0 >> 9;  // 512 rows per instance

  float g[4] = {1.f, 0.f, 0.f, 0.f};
  if (MOE) {
#pragma unroll
    for (int e = 0; e < 4; ++e) g[e] = gw[inst * 4 + e];
  }

  v8f acc[4][2];
  {
    v8f z = {};
#pragma unroll
    for (int i = 0; i < 4; ++i)
#pragma unroll
      for (int j = 0; j < 2; ++j) acc[i][j] = z;
  }

  const int mw = (wave >> 2) * 64;  // 0 or 64
  const int nw = (wave & 3) * 32;   // 0,32,64,96

  for (int kb = 0; kb < KDIM; kb += 32) {
    __syncthreads();

    // ---- A tile 128x32: 512 chunks of 8 halves (16B), 2 per thread ----
#pragma unroll
    for (int it = 0; it < 2; ++it) {
      int chunk = tid + it * 256;
      int r = chunk >> 2, c8 = (chunk & 3) * 8;
      _Float16* src = const_cast<_Float16*>(&A[(size_t)(row0 + r) * KDIM + kb + c8]);
      _Float16* dst = &As[r * 40 + c8];
#ifdef ASYNC_LDS
      __builtin_amdgcn_global_load_async_to_lds_b128(
          (gas_v4i_ptr)src, (las_v4i_ptr)dst, 0, 0);
#else
      *(v8h*)dst = *(const v8h*)src;
#endif
    }

    // prefetch next K-step's weight tiles (L2-resident; hides latency)
    if (kb + 32 < KDIM) {
      const float* p = &W[(size_t)(kb + 32) * NDIM + n0 + (tid & 127)];
      for (int e = 0; e < E; ++e)
        __builtin_prefetch(p + (size_t)e * KDIM * NDIM, 0, 0);
    }

    // ---- W tile 32x128: gate-combine E experts, store transposed f16 ----
    // 1024 groups of 4 floats, 4 per thread (overlaps with async A staging)
#pragma unroll
    for (int it = 0; it < 4; ++it) {
      int grp = tid + it * 256;
      int k = grp >> 5, n4 = (grp & 31) * 4;
      v4f a4 = {0.f, 0.f, 0.f, 0.f};
      for (int e = 0; e < E; ++e) {
        v4f w4 = *(const v4f*)&W[(size_t)e * KDIM * NDIM +
                                 (size_t)(kb + k) * NDIM + n0 + n4];
        a4 += g[e] * w4;
      }
#pragma unroll
      for (int j = 0; j < 4; ++j) Ws[(n4 + j) * 40 + k] = (_Float16)a4[j];
    }

    wait_asynccnt0();
    __syncthreads();

    v16h af[4], bf[2];
#pragma unroll
    for (int i = 0; i < 4; ++i)
      af[i] = frag_ld(&As[(mw + i * 16 + rl) * 40 + sel8]);
#pragma unroll
    for (int j = 0; j < 2; ++j)
      bf[j] = frag_ld(&Ws[(nw + j * 16 + rl) * 40 + sel8]);
#pragma unroll
    for (int i = 0; i < 4; ++i)
#pragma unroll
      for (int j = 0; j < 2; ++j) acc[i][j] = wmma32(af[i], bf[j], acc[i][j]);
  }

  // epilogue: D layout = lane%16 -> col, vgpr r + 8*(lane/16) -> row
#pragma unroll
  for (int i = 0; i < 4; ++i)
#pragma unroll
    for (int j = 0; j < 2; ++j)
#pragma unroll
      for (int r = 0; r < 8; ++r) {
        int rg = row0 + mw + i * 16 + r + (lane >> 4) * 8;
        int cg = n0 + nw + j * 16 + rl;
        float b = MOE ? bias[inst * NDIM + cg] : bias[cg];
        Y[(size_t)rg * NDIM + cg] = (OutT)(acc[i][j][r] + b);
      }
}

// --------------------------- 4) span attention -----------------------------
// grid = (S/64 qtiles, H, NINST), block = 128 (4 waves, 16 q-rows each).
// LDS: scores/probs f16 [64][512], V^T f16 [64][520], rowinv f32 [64].
#define ATTN_SMEM (64 * 512 * 2 + 64 * 520 * 2 + 64 * 4)

__global__ void span_attn(const _Float16* __restrict__ Q,
                          const _Float16* __restrict__ Kx,
                          const _Float16* __restrict__ V,
                          _Float16* __restrict__ O) {
  extern __shared__ char smem_raw[];
  _Float16* sc = (_Float16*)smem_raw;     // [64][512] scores -> probs in place
  _Float16* vt = sc + 64 * 512;           // [64][520] V transposed [dh][key]
  float* rinv = (float*)(vt + 64 * 520);  // [64]

  const int tid = threadIdx.x;
  const int lane = tid & 31, wave = tid >> 5;
  const int rl = lane & 15;
  const int sel8 = (lane >> 4) * 8;
  const int q0 = blockIdx.x * 64;
  const int h = blockIdx.y;
  const int inst = blockIdx.z;
  const size_t base = (size_t)inst * SPAN * 1024 + h * 64;

  // stage V^T: 16B global reads along Dh, scatter-transpose into LDS
  for (int chunk = tid; chunk < SPAN * 8; chunk += 128) {
    int key = chunk >> 3, d8 = (chunk & 7) * 8;
    v8h v = *(const v8h*)&V[base + (size_t)key * 1024 + d8];
#pragma unroll
    for (int j = 0; j < 8; ++j) vt[(d8 + j) * 520 + key] = v[j];
  }
  __syncthreads();

  // Q fragments for this wave's 16 rows (K=Dh=64 -> 2 k-steps), loaded once
  v16h qa[2];
#pragma unroll
  for (int t = 0; t < 2; ++t)
    qa[t] = frag_ld(&Q[base + (size_t)(q0 + wave * 16 + rl) * 1024 + t * 32 + sel8]);

  // scores = (Q K^T) * 1/sqrt(Dh); B-frag of K^T is contiguous per lane
  for (int kt = 0; kt < 32; ++kt) {
    v8f s = {};
#pragma unroll
    for (int t = 0; t < 2; ++t) {
      v16h kb = frag_ld(&Kx[base + (size_t)(kt * 16 + rl) * 1024 + t * 32 + sel8]);
      s = wmma32(qa[t], kb, s);
    }
#pragma unroll
    for (int r = 0; r < 8; ++r)
      sc[(wave * 16 + r + (lane >> 4) * 8) * 512 + kt * 16 + rl] =
          (_Float16)(s[r] * 0.125f);
  }

  // full-row softmax (each wave owns its 16 rows; no cross-wave barrier)
  for (int r = 0; r < 16; ++r) {
    int row = wave * 16 + r;
    float mx = -3.0e38f;
    for (int c = lane; c < 512; c += 32)
      mx = fmaxf(mx, (float)sc[row * 512 + c]);
#pragma unroll
    for (int off = 16; off > 0; off >>= 1)
      mx = fmaxf(mx, __shfl_xor(mx, off, 32));
    float sum = 0.f;
    for (int c = lane; c < 512; c += 32) {
      float ev = __expf((float)sc[row * 512 + c] - mx);
      sc[row * 512 + c] = (_Float16)ev;  // in-place probs
      sum += ev;
    }
#pragma unroll
    for (int off = 16; off > 0; off >>= 1) sum += __shfl_xor(sum, off, 32);
    if (lane == 0) rinv[row] = 1.0f / sum;
  }

  // O = P V  (K dim = 512 keys -> 16 k-steps; V^T gives contiguous B-frags)
  v8f oacc[4];
  {
    v8f z = {};
#pragma unroll
    for (int n = 0; n < 4; ++n) oacc[n] = z;
  }
  for (int t = 0; t < 16; ++t) {
    v16h pa = frag_ld(&sc[(wave * 16 + rl) * 512 + t * 32 + sel8]);
#pragma unroll
    for (int n = 0; n < 4; ++n) {
      v16h vb = frag_ld(&vt[(n * 16 + rl) * 520 + t * 32 + sel8]);
      oacc[n] = wmma32(pa, vb, oacc[n]);
    }
  }

#pragma unroll
  for (int n = 0; n < 4; ++n)
#pragma unroll
    for (int r = 0; r < 8; ++r) {
      int rowl = wave * 16 + r + (lane >> 4) * 8;
      float inv = rinv[rowl];
      O[base + (size_t)(q0 + rowl) * 1024 + n * 16 + rl] =
          (_Float16)(oacc[n][r] * inv);
    }
}

// ------------------------------- launcher ----------------------------------
extern "C" void kernel_launch(void* const* d_in, const int* in_sizes, int n_in,
                              void* d_out, int out_size, void* d_ws,
                              size_t ws_size, hipStream_t stream) {
  const float* X   = (const float*)d_in[0];
  const float* Wq  = (const float*)d_in[1];
  const float* bq  = (const float*)d_in[2];
  const float* Wk  = (const float*)d_in[3];
  const float* bk  = (const float*)d_in[4];
  const float* Wv  = (const float*)d_in[5];
  const float* bv  = (const float*)d_in[6];
  const float* gW1 = (const float*)d_in[7];
  const float* gb1 = (const float*)d_in[8];
  const float* gW2 = (const float*)d_in[9];
  const float* gb2 = (const float*)d_in[10];
  const float* Wo  = (const float*)d_in[11];
  const float* bo  = (const float*)d_in[12];

  char* ws = (char*)d_ws;
  _Float16* Xh = (_Float16*)(ws + OFF_XH);
  _Float16* Qb = (_Float16*)(ws + OFF_Q);
  _Float16* Kb = (_Float16*)(ws + OFF_K);
  _Float16* Vb = (_Float16*)(ws + OFF_V);
  _Float16* AO = (_Float16*)(ws + OFF_AO);
  float* gw    = (float*)(ws + OFF_GW);
  float* beff  = (float*)(ws + OFF_BE);  // [3][16][1024]

  const int total = MDIM * KDIM;
  cvt_f32_to_f16<<<2048, 256, 0, stream>>>(X, Xh, total);

  gate_kernel<<<NINST, 256, 0, stream>>>(X, gW1, gb1, gW2, gb2, bq, bk, bv, gw,
                                         beff);

  dim3 ggrid(NDIM / 128, MDIM / 128);
  moe_gemm<_Float16, true><<<ggrid, 256, 0, stream>>>(
      Xh, Wq, gw, beff + 0 * NINST * NDIM, Qb, 4);
  moe_gemm<_Float16, true><<<ggrid, 256, 0, stream>>>(
      Xh, Wk, gw, beff + 1 * NINST * NDIM, Kb, 4);
  moe_gemm<_Float16, true><<<ggrid, 256, 0, stream>>>(
      Xh, Wv, gw, beff + 2 * NINST * NDIM, Vb, 4);

  dim3 agrid(SPAN / 64, 16, NINST);
  span_attn<<<agrid, 128, ATTN_SMEM, stream>>>(Qb, Kb, Vb, AO);

  moe_gemm<float, false><<<ggrid, 256, 0, stream>>>(AO, Wo, nullptr, bo,
                                                    (float*)d_out, 1);
}
